// MuSc_10462540333176
// MI455X (gfx1250) — matrix operand order repaired and checked by reading, not compiled, not run on hardware
//
#include <hip/hip_runtime.h>
#include <hip/hip_bf16.h>

typedef __attribute__((ext_vector_type(16))) _Float16 v16h;
typedef __attribute__((ext_vector_type(8)))  _Float16 v8h;
typedef __attribute__((ext_vector_type(4)))  _Float16 v4h;
typedef __attribute__((ext_vector_type(8)))  float    v8f;

#define LL 4       // layers
#define NB 8       // batch images
#define PH_ 24
#define PW_ 24
#define PP 576     // patches per image
#define DD 1024    // feature dim
#define HH 336     // output H=W

// GEMM tiling
#define WGM 192            // workgroup tile M (rows, image bi)
#define WGN 192            // workgroup tile N (cols, image bj)
#define KC 64              // K chunk staged in LDS (2 wmma k-steps)
#define LDST 72            // LDS row stride in halves (64 + 8 pad) = 144B, conflict-free
#define BUFH ((WGM + WGN) * LDST)   // halves per LDS buffer = 27648 (55296 B)

// ---------------- stage 1: r x r avg pool (zero-pad, /r^2) + L2 norm -> f16
__global__ __launch_bounds__(256)
void pool_norm_kernel(const float* __restrict__ feat, _Float16* __restrict__ X, int r) {
  int blk = blockIdx.x;                 // l*NB*PP + b*PP + p
  int l = blk / (NB * PP);
  int rem = blk % (NB * PP);
  int b = rem / PP;
  int p = rem % PP;
  int py = p / PW_, px = p % PW_;
  int t = threadIdx.x;
  int d = t * 4;                        // 1024 / 256 = 4 channels per thread
  float sx = 0.f, sy = 0.f, sz = 0.f, sw = 0.f;
  int pad = r >> 1;
  const float* base = feat + (((size_t)l * NB + b) * PP) * DD;
  for (int dy = -pad; dy <= pad; ++dy) {
    int y = py + dy;
    if (y < 0 || y >= PH_) continue;
    for (int dx = -pad; dx <= pad; ++dx) {
      int x = px + dx;
      if (x < 0 || x >= PW_) continue;
      const float4 v = *(const float4*)(base + (size_t)(y * PW_ + x) * DD + d);
      sx += v.x; sy += v.y; sz += v.z; sw += v.w;
    }
  }
  float inv_rr = 1.0f / (float)(r * r);
  sx *= inv_rr; sy *= inv_rr; sz *= inv_rr; sw *= inv_rr;
  float local = sx * sx + sy * sy + sz * sz + sw * sw;
  __shared__ float red[256];
  red[t] = local;
  __syncthreads();
  for (int st = 128; st > 0; st >>= 1) {
    if (t < st) red[t] += red[t + st];
    __syncthreads();
  }
  float inv = 1.0f / sqrtf(red[0]);
  _Float16* out = X + (((size_t)l * NB + b) * PP + p) * DD + d;
  v4h o;
  o.x = (_Float16)(sx * inv); o.y = (_Float16)(sy * inv);
  o.z = (_Float16)(sz * inv); o.w = (_Float16)(sw * inv);
  *(v4h*)out = o;
}

// ---------------- init helpers ----------------
__global__ void fill_u32(unsigned int* p, unsigned int v, int n) {
  int i = blockIdx.x * blockDim.x + threadIdx.x;
  if (i < n) p[i] = v;
}
__global__ void zero_f32(float* p, int n) {
  int i = blockIdx.x * blockDim.x + threadIdx.x;
  if (i < n) p[i] = 0.0f;
}

// ---------------- stage 2: LDS-staged WMMA GEMM S = X Xt + fused min-dist ---
// 384 threads = 12 waves (4 M x 3 N), wave tile 48x64, WG tile 192x192.
// Double-buffered KC=64 chunks loaded with global_load_async_to_lds_b128.
__global__ __launch_bounds__(384)
void msm_gemm(const _Float16* __restrict__ X, unsigned int* __restrict__ dmin) {
  extern __shared__ _Float16 smem[];    // 2 * BUFH halves = 110592 bytes (dynamic)

  const int tid  = threadIdx.x;
  const int wid  = tid >> 5;            // 0..11
  const int lane = tid & 31;
  const int lane16 = lane & 15;
  const int g = (lane >> 4) & 1;
  const int wm = wid / 3;               // 0..3  (M position)
  const int wn = wid % 3;               // 0..2  (N position)

  // block decode: 4 layers * 28 pairs * 9 (3x3) WG tiles = 1008 blocks
  int bidx = blockIdx.x;
  int l = bidx / 252;
  int rem = bidx % 252;
  int pairIdx = rem / 9;
  int tq = rem % 9;
  int tmWG = tq / 3, tnWG = tq % 3;
  int bi = 0, pr = pairIdx, cnt = NB - 1;
  while (pr >= cnt) { pr -= cnt; ++bi; --cnt; }
  int bj = bi + 1 + pr;

  const _Float16* Xl = X + (size_t)l * (NB * PP * DD);
  const int m0 = bi * PP + tmWG * WGM;  // first global row of A tile
  const int n0 = bj * PP + tnWG * WGN;  // first global row of B tile

  // ---- async-copy address precompute: 16B units; A: 192 rows * 8 units/row
  unsigned ldsbase = (unsigned)(size_t)(void*)smem;  // LDS byte offset of array
  unsigned lA[4], vA[4], lB[4], vB[4];
#pragma unroll
  for (int s = 0; s < 4; ++s) {
    int u = tid + 384 * s;              // 0..1535
    int row = u >> 3, q = u & 7;        // q*8 halves = q*16 bytes into the row
    lA[s] = (unsigned)((row * LDST + q * 8) * 2);
    vA[s] = (unsigned)((((m0 + row) * DD) + q * 8) * 2);
    lB[s] = (unsigned)(((WGM + row) * LDST + q * 8) * 2);
    vB[s] = (unsigned)((((n0 + row) * DD) + q * 8) * 2);
  }

  auto issue_chunk = [&](int k0, int buf) {
    unsigned bbase = ldsbase + (unsigned)(buf * BUFH * 2);
    unsigned koff = (unsigned)(k0 * 2);
#pragma unroll
    for (int s = 0; s < 4; ++s) {
      unsigned lo = bbase + lA[s];
      unsigned vo = vA[s] + koff;
      asm volatile("global_load_async_to_lds_b128 %0, %1, %2"
                   :: "v"(lo), "v"(vo), "s"(Xl) : "memory");
    }
#pragma unroll
    for (int s = 0; s < 4; ++s) {
      unsigned lo = bbase + lB[s];
      unsigned vo = vB[s] + koff;
      asm volatile("global_load_async_to_lds_b128 %0, %1, %2"
                   :: "v"(lo), "v"(vo), "s"(Xl) : "memory");
    }
  };

  v8f acc[3][4] = {};
  issue_chunk(0, 0);

  for (int c = 0; c < DD / KC; ++c) {           // 16 chunks
    if (c < DD / KC - 1) {
      issue_chunk((c + 1) * KC, (c + 1) & 1);
      asm volatile("s_wait_asynccnt 8" ::: "memory");   // chunk c complete
    } else {
      asm volatile("s_wait_asynccnt 0" ::: "memory");
    }
    __syncthreads();

    const _Float16* Ab = smem + (size_t)((c & 1) * BUFH);
    const _Float16* Bb = Ab + WGM * LDST;
#pragma unroll
    for (int ks = 0; ks < KC; ks += 32) {
      v16h a[3], b[4];
#pragma unroll
      for (int i = 0; i < 3; ++i) {
        const _Float16* rp = Ab + (wm * 48 + i * 16 + lane16) * LDST + ks;
        union { v16h v; v8h h[2]; } u;
        u.h[0] = *(const v8h*)(rp + g * 8);
        u.h[1] = *(const v8h*)(rp + 16 + g * 8);
        a[i] = u.v;
      }
#pragma unroll
      for (int j = 0; j < 4; ++j) {
        const _Float16* rp = Bb + (wn * 64 + j * 16 + lane16) * LDST + ks + g * 8;
        union { v16h v; v8h h[2]; } u;
        u.h[0] = *(const v8h*)(rp);
        u.h[1] = *(const v8h*)(rp + 16);
        b[j] = u.v;
      }
#pragma unroll
      for (int i = 0; i < 3; ++i)
#pragma unroll
        for (int j = 0; j < 4; ++j)
          acc[i][j] = __builtin_amdgcn_wmma_f32_16x16x32_f16(
              false, a[i], false, b[j], (short)0, acc[i][j], false, false);
    }
    __syncthreads();   // all waves done reading before buffer is overwritten
  }

  // min distance == max dot.  d2 = clamp(2 - 2*maxdot, >=0); uint bits order as float.
  // ---- row direction: queries = image bi rows (min over this wave's 64 n)
#pragma unroll
  for (int i = 0; i < 3; ++i) {
#pragma unroll
    for (int v = 0; v < 8; ++v) {
      float x = fmaxf(fmaxf(acc[i][0][v], acc[i][1][v]), fmaxf(acc[i][2][v], acc[i][3][v]));
      x = fmaxf(x, __shfl_xor(x, 8, 16));
      x = fmaxf(x, __shfl_xor(x, 4, 16));
      x = fmaxf(x, __shfl_xor(x, 2, 16));
      x = fmaxf(x, __shfl_xor(x, 1, 16));
      if (lane16 == 0) {
        int pm = tmWG * WGM + wm * 48 + i * 16 + v + 8 * g;  // C row M = v + 8*(lane/16)
        float d2 = fmaxf(0.0f, 2.0f - 2.0f * x);
        atomicMin(&dmin[(((size_t)l * NB + bi) * PP + pm) * NB + bj], __float_as_uint(d2));
      }
    }
  }
  // ---- col direction: queries = image bj cols (min over this wave's 48 m)
#pragma unroll
  for (int j = 0; j < 4; ++j) {
    float x = acc[0][j][0];
#pragma unroll
    for (int v = 1; v < 8; ++v) x = fmaxf(x, acc[0][j][v]);
#pragma unroll
    for (int v = 0; v < 8; ++v) x = fmaxf(x, fmaxf(acc[1][j][v], acc[2][j][v]));
    x = fmaxf(x, __shfl_xor(x, 16, 32));   // combine M = v and M = v+8 halves
    if (lane < 16) {
      int pn = tnWG * WGN + wn * 64 + j * 16 + lane16;
      float d2 = fmaxf(0.0f, 2.0f - 2.0f * x);
      atomicMin(&dmin[(((size_t)l * NB + bj) * PP + pn) * NB + bi], __float_as_uint(d2));
    }
  }
}

// ---------------- stage 3: mean of 2 smallest dists, accumulate over l,r ----
__global__ void stage2_scores(const unsigned int* __restrict__ dmin, float* __restrict__ scores) {
  int idx = blockIdx.x * blockDim.x + threadIdx.x;   // 0..4607
  if (idx >= NB * PP) return;
  int b = idx / PP, p = idx % PP;
  float accum = 0.0f;
  for (int l = 0; l < LL; ++l) {
    const unsigned int* row = dmin + (((size_t)l * NB + b) * PP + p) * NB;
    float m1 = __builtin_inff(), m2 = __builtin_inff();   // self stays +inf -> excluded
    for (int c = 0; c < NB; ++c) {
      float v = __uint_as_float(row[c]);
      if (v < m1) { m2 = m1; m1 = v; }
      else if (v < m2) { m2 = v; }
    }
    accum += 0.5f * (sqrtf(m1) + sqrtf(m2));       // mean of ds[0:2]
  }
  scores[idx] += accum * (1.0f / (LL * 3.0f));     // mean over l then mean over r
}

// ---------------- finalize ----------------
__global__ void image_max_kernel(const float* __restrict__ scores, float* __restrict__ out) {
  int b = blockIdx.x;
  int t = threadIdx.x;
  float m = -__builtin_inff();
  for (int p = t; p < PP; p += 256) m = fmaxf(m, scores[b * PP + p]);
  __shared__ float red[256];
  red[t] = m;
  __syncthreads();
  for (int st = 128; st > 0; st >>= 1) {
    if (t < st) red[t] = fmaxf(red[t], red[t + st]);
    __syncthreads();
  }
  if (t == 0) out[b] = red[0];
}

__global__ void bilinear_kernel(const float* __restrict__ scores, float* __restrict__ out) {
  int idx = blockIdx.x * blockDim.x + threadIdx.x;   // < 8*336*336
  int b = idx / (HH * HH);
  int rem = idx % (HH * HH);
  int y = rem / HH, x = rem % HH;
  const float sc = 23.0f / 335.0f;                   // (PH-1)/(H-1), align corners
  float ys = y * sc, xs = x * sc;
  int y0 = (int)ys, x0 = (int)xs;
  int y1 = min(y0 + 1, PH_ - 1), x1 = min(x0 + 1, PW_ - 1);
  float wy = ys - (float)y0, wx = xs - (float)x0;
  const float* S = scores + b * PP;
  float f00 = S[y0 * PW_ + x0], f01 = S[y0 * PW_ + x1];
  float f10 = S[y1 * PW_ + x0], f11 = S[y1 * PW_ + x1];
  out[idx] = f00 * (1 - wy) * (1 - wx) + f01 * (1 - wy) * wx +
             f10 * wy * (1 - wx) + f11 * wy * wx;
}

// ---------------- host orchestration ----------------
extern "C" void kernel_launch(void* const* d_in, const int* in_sizes, int n_in,
                              void* d_out, int out_size, void* d_ws, size_t ws_size,
                              hipStream_t stream) {
  (void)in_sizes; (void)n_in; (void)out_size; (void)ws_size;
  const float* feat = (const float*)d_in[0];
  char* ws = (char*)d_ws;

  const size_t X_BYTES = (size_t)LL * NB * PP * DD * 2;          // 37,748,736
  const size_t DMIN_N  = (size_t)LL * NB * PP * NB;              // 147,456
  _Float16*     X      = (_Float16*)ws;
  unsigned int* dmin   = (unsigned int*)(ws + X_BYTES);
  float*        scores = (float*)(ws + X_BYTES + DMIN_N * 4);
  float*        out    = (float*)d_out;

  zero_f32<<<(NB * PP + 255) / 256, 256, 0, stream>>>(scores, NB * PP);

  const int radii[3] = {1, 3, 5};
  const int gemm_lds = 2 * BUFH * 2;   // 110592 bytes dynamic LDS
  for (int ri = 0; ri < 3; ++ri) {
    pool_norm_kernel<<<LL * NB * PP, 256, 0, stream>>>(feat, X, radii[ri]);
    fill_u32<<<(int)((DMIN_N + 255) / 256), 256, 0, stream>>>(dmin, 0x7F800000u, (int)DMIN_N);
    // 4 layers * 28 pairs * 9 WG tiles = 1008 blocks of 12 waves
    msm_gemm<<<1008, 384, gemm_lds, stream>>>(X, dmin);
    stage2_scores<<<(NB * PP + 255) / 256, 256, 0, stream>>>(dmin, scores);
  }

  image_max_kernel<<<NB, 256, 0, stream>>>(scores, out);
  bilinear_kernel<<<(NB * HH * HH) / 256, 256, 0, stream>>>(scores, out + NB);
}
